// IntegrationMeasure_19894288515165
// MI455X (gfx1250) — matrix operand order repaired and checked by reading, not compiled, not run on hardware
//
#include <hip/hip_runtime.h>
#include <hip/hip_bf16.h>
#include <math.h>

// ---------------------------------------------------------------------------
// IntegrationMeasure (phi) for MI455X / gfx1250, wave32 + WMMA bf16.
//
// Roofline: 172 GFLOP vs 336 MB HBM traffic -> AI ~512 FLOP/B. f32 WMMA
// (16x16x4) would be compute bound; bf16 WMMA (16x16x32, f32 accum) is 8x
// denser and approaches the 23.3 TB/s HBM roof. Activations are streamed
// from HBM exactly once (each block covers full N); weights are converted
// to bf16 once and served from the 192 MB L2.
//
// Workspace layout (d_ws, needs 8 MB + 32 B):
//   [0 .. 4MB)   : Ww as bf16  (1024 x 2048)
//   [4MB .. 8MB) : Wp as bf16  (1024 x 2048)
//   [8MB .. +32B): 8 float accumulators (4 whole-norm sums, 4 parts-norm sums)
// ---------------------------------------------------------------------------

typedef __attribute__((ext_vector_type(16))) __bf16 v16bf;
typedef __attribute__((ext_vector_type(8)))  __bf16 v8bf;
typedef __attribute__((ext_vector_type(4)))  __bf16 v4bf;
typedef __attribute__((ext_vector_type(8)))  float  v8f;

#define D_MODEL 2048
#define D_HALF  1024
#define BATCH   4
#define SEQ     2048
#define HIST    4

#define M_T     32                    // rows per workgroup
#define KC      512                   // K-chunk staged in LDS (bf16)
#define NKCH    (D_MODEL / KC)        // 4 chunks
#define KSTEP   32                    // WMMA K per instruction
#define NKK     (KC / KSTEP)          // 16 wmma k-steps per chunk
#define LDS_PAD 8                     // bf16 elems padding -> bank-conflict-free
#define ROWLEN  (KC + LDS_PAD)

#define WHOLE_BLOCKS (BATCH * SEQ / M_T)          // 256
#define PARTS_BLOCKS (HIST * BATCH * SEQ / M_T)   // 1024

// ---------------- weight f32 -> bf16 conversion (one cheap pass) ------------
__global__ void convert_weights_kernel(const float* __restrict__ Ww,
                                       const float* __restrict__ Wp,
                                       __bf16* __restrict__ Wwb,
                                       __bf16* __restrict__ Wpb) {
  const int i  = blockIdx.x * blockDim.x + threadIdx.x;   // float4 index
  const int n4 = D_HALF * D_MODEL / 4;
  if (i >= n4) return;
  float4 a = ((const float4*)Ww)[i];
  float4 b = ((const float4*)Wp)[i];
  v4bf av = { (__bf16)a.x, (__bf16)a.y, (__bf16)a.z, (__bf16)a.w };
  v4bf bv = { (__bf16)b.x, (__bf16)b.y, (__bf16)b.z, (__bf16)b.w };
  ((v4bf*)Wwb)[i] = av;
  ((v4bf*)Wpb)[i] = bv;
}

__global__ void zero_accum_kernel(float* __restrict__ accum) {
  if (threadIdx.x < 2 * BATCH) accum[threadIdx.x] = 0.0f;
}

// ---------------- main fused GEMM + row-norm kernel -------------------------
__launch_bounds__(256)
__global__ void phi_norm_gemm_kernel(const float* __restrict__ Xcur,
                                     const float* __restrict__ Xhist,
                                     const __bf16* __restrict__ Wwb,
                                     const __bf16* __restrict__ Wpb,
                                     const float* __restrict__ bw,
                                     const float* __restrict__ bp,
                                     float* __restrict__ accum) {
  __shared__ __bf16 As[M_T][ROWLEN];   // 32 x 520 bf16 = 33,280 B
  __shared__ float  red[8][M_T];       // per-wave per-row partial sumsq

  const int tid  = threadIdx.x;
  const int wave = tid >> 5;           // 8 wave32s
  const int lane = tid & 31;
  const int ln   = lane & 15;          // position within 16-lane half
  const int hf   = lane >> 4;          // which 16-lane half

  // ---- block -> (source rows, weight matrix, bias, output accumulator) ----
  const float*  X;
  const __bf16* W;
  const float*  bias;
  float*        accp;
  {
    int bid = blockIdx.x;
    if (bid < WHOLE_BLOCKS) {
      int b  = bid / (SEQ / M_T);
      int st = bid % (SEQ / M_T);
      X    = Xcur + ((size_t)b * SEQ + (size_t)st * M_T) * D_MODEL;
      W    = Wwb;  bias = bw;  accp = accum + b;
    } else {
      int t  = bid - WHOLE_BLOCKS;
      int hb = t / (SEQ / M_T);        // flat h*BATCH + b (history is [h][b][s][d])
      int st = t % (SEQ / M_T);
      int b  = hb & (BATCH - 1);
      X    = Xhist + ((size_t)hb * SEQ + (size_t)st * M_T) * D_MODEL;
      W    = Wpb;  bias = bp;  accp = accum + BATCH + b;
    }
  }

  // f32 accumulators: 2 M-tiles x 8 N-tiles per wave = 128 VGPRs
  v8f acc[2][8];
#pragma unroll
  for (int m = 0; m < 2; ++m)
#pragma unroll
    for (int n = 0; n < 8; ++n)
      acc[m][n] = (v8f)(0.0f);

  for (int kc = 0; kc < NKCH; ++kc) {
    __syncthreads();   // previous chunk fully consumed before restage
    // ---- stage 32 x 512 f32 chunk into LDS as bf16 (each elem read once) --
    for (int i = tid; i < M_T * KC / 4; i += 256) {
      int r  = i / (KC / 4);
      int c4 = i % (KC / 4);
      float4 f = *(const float4*)(X + (size_t)r * D_MODEL + kc * KC + c4 * 4);
      v4bf v = { (__bf16)f.x, (__bf16)f.y, (__bf16)f.z, (__bf16)f.w };
      *(v4bf*)&As[r][c4 * 4] = v;
    }
    __syncthreads();

    for (int kk = 0; kk < NKK; ++kk) {
      const int kb = kk * KSTEP;
      // A fragment (16x32 bf16, contiguous K order matching B): lane ln, half hf holds
      // K = [kb+hf*16 .. +15] -> two 16B LDS loads.
      v8bf a0l = *(const v8bf*)&As[ln][kb + hf * 16];
      v8bf a0h = *(const v8bf*)&As[ln][kb + hf * 16 + 8];
      v8bf a1l = *(const v8bf*)&As[16 + ln][kb + hf * 16];
      v8bf a1h = *(const v8bf*)&As[16 + ln][kb + hf * 16 + 8];
      v16bf a0 = __builtin_shufflevector(a0l, a0h, 0,1,2,3,4,5,6,7,8,9,10,11,12,13,14,15);
      v16bf a1 = __builtin_shufflevector(a1l, a1h, 0,1,2,3,4,5,6,7,8,9,10,11,12,13,14,15);
#pragma unroll
      for (int ntl = 0; ntl < 8; ++ntl) {
        // B fragment (32x16 bf16): lane's column N = ncol, 16 contiguous
        // bf16 of weight row at K = kb + hf*16 (weights stored row-major [N][K]).
        const int ncol = (wave * 8 + ntl) * 16 + ln;
        const __bf16* wp = W + (size_t)ncol * D_MODEL + (size_t)(kc * KC + kb + hf * 16);
        v16bf bfrag = *(const v16bf*)wp;
        acc[0][ntl] = __builtin_amdgcn_wmma_f32_16x16x32_bf16(
            false, a0, false, bfrag, (short)0, acc[0][ntl], false, false);
        acc[1][ntl] = __builtin_amdgcn_wmma_f32_16x16x32_bf16(
            false, a1, false, bfrag, (short)0, acc[1][ntl], false, false);
      }
    }
  }

  // ---- epilogue: (+bias)^2, reduce over N, sqrt per row, mean via atomics --
  float sq0[8], sq1[8];
#pragma unroll
  for (int v = 0; v < 8; ++v) { sq0[v] = 0.0f; sq1[v] = 0.0f; }

#pragma unroll
  for (int ntl = 0; ntl < 8; ++ntl) {
    const int ncol = (wave * 8 + ntl) * 16 + ln;
    const float bv = bias[ncol];
#pragma unroll
    for (int v = 0; v < 8; ++v) {
      float t0 = acc[0][ntl][v] + bv;  sq0[v] += t0 * t0;
      float t1 = acc[1][ntl][v] + bv;  sq1[v] += t1 * t1;
    }
  }
  // sum across the 16 lanes of each half (xor masks <=8 stay inside a half;
  // C/D layout: VGPR v, half hf -> row v + 8*hf, lane%16 -> column).
#pragma unroll
  for (int mask = 1; mask <= 8; mask <<= 1) {
#pragma unroll
    for (int v = 0; v < 8; ++v) {
      sq0[v] += __shfl_xor(sq0[v], mask, 32);
      sq1[v] += __shfl_xor(sq1[v], mask, 32);
    }
  }
  if (ln == 0) {
#pragma unroll
    for (int v = 0; v < 8; ++v) {
      red[wave][hf * 8 + v]      = sq0[v];   // rows 0..15  (M-tile 0)
      red[wave][16 + hf * 8 + v] = sq1[v];   // rows 16..31 (M-tile 1)
    }
  }
  __syncthreads();

  if (tid < M_T) {
    float s = 0.0f;
#pragma unroll
    for (int w = 0; w < 8; ++w) s += red[w][tid];  // cross-wave N reduction
    float norm = sqrtf(s);
#pragma unroll
    for (int mask = 1; mask < 32; mask <<= 1)
      norm += __shfl_xor(norm, mask, 32);          // sum 32 row norms
    if (tid == 0) atomicAdd(accp, norm);           // one atomic per block
  }
}

// ---------------- finalize: phi per batch -----------------------------------
__global__ void finalize_kernel(const float* __restrict__ accum,
                                const float* __restrict__ phi_scale,
                                const float* __restrict__ phi_bias,
                                float* __restrict__ out) {
  const int b = threadIdx.x;
  if (b < BATCH) {
    float w   = accum[b]         * (1.0f / (float)SEQ);
    float p   = accum[BATCH + b] * (1.0f / (float)(SEQ * HIST));
    float raw = (w - p) / (w + 1e-8f);
    float phi = phi_scale[0] * raw + phi_bias[0];
    out[b] = fminf(fmaxf(phi, 0.0f), 1.0f);
  }
}

// ---------------------------------------------------------------------------
extern "C" void kernel_launch(void* const* d_in, const int* in_sizes, int n_in,
                              void* d_out, int out_size, void* d_ws, size_t ws_size,
                              hipStream_t stream) {
  const float* Xcur      = (const float*)d_in[0];   // [4,2048,2048]
  const float* Xhist     = (const float*)d_in[1];   // [4,4,2048,2048]
  const float* Ww        = (const float*)d_in[2];   // [1024,2048]
  const float* bw        = (const float*)d_in[3];   // [1024]
  const float* Wp        = (const float*)d_in[4];   // [1024,2048]
  const float* bp        = (const float*)d_in[5];   // [1024]
  const float* phi_scale = (const float*)d_in[6];   // scalar
  const float* phi_bias  = (const float*)d_in[7];   // scalar
  float* out = (float*)d_out;                       // [4]

  const size_t wbytes = (size_t)D_HALF * D_MODEL * sizeof(__bf16);  // 4 MB
  __bf16* Wwb  = (__bf16*)d_ws;
  __bf16* Wpb  = (__bf16*)((char*)d_ws + wbytes);
  float*  accp = (float*)((char*)d_ws + 2 * wbytes);

  const int n4 = D_HALF * D_MODEL / 4;  // float4 count per weight matrix
  convert_weights_kernel<<<(n4 + 255) / 256, 256, 0, stream>>>(Ww, Wp, Wwb, Wpb);
  zero_accum_kernel<<<1, 32, 0, stream>>>(accp);
  phi_norm_gemm_kernel<<<WHOLE_BLOCKS + PARTS_BLOCKS, 256, 0, stream>>>(
      Xcur, Xhist, Wwb, Wpb, bw, bp, accp);
  finalize_kernel<<<1, 32, 0, stream>>>(accp, phi_scale, phi_bias, out);
}
